// SymmetryAttention_9191230013549
// MI455X (gfx1250) — matrix operand order repaired and checked by reading, not compiled, not run on hardware
//
#include <hip/hip_runtime.h>
#include <hip/hip_bf16.h>

typedef __attribute__((ext_vector_type(16))) _Float16 v16h;
typedef __attribute__((ext_vector_type(8)))  _Float16 v8h;
typedef __attribute__((ext_vector_type(4)))  _Float16 v4h;
typedef __attribute__((ext_vector_type(8)))  float    v8f;

#define NB    2
#define CIN   1088
#define CMID  136
#define CP    160          // CMID padded to 5 * 32 (zero-padded K for energy GEMM)
#define HW    4096
#define EPS_IN 1e-5f
#define ALPHA  100.0f
#define SLOPE  0.2f

// ---------------------------------------------------------------------------
// CDNA5 WMMA fragment loaders (wave32, v_wmma_f32_16x16x32_f16 layouts).
// A (16x32 f16): lane L<16 -> row L,  K in {0..7, 16..23}
//                lane L>=16 -> row L-16, K in {8..15, 24..31}
// B (32x16 f16): lane L<16 -> col L,  K in {0..7, 16..23}
//                lane L>=16 -> col L-16, K in {8..15, 24..31}
// C/D (16x16 f32, 8 VGPRs): lane L: col = L&15, row = v + (L>=16 ? 8 : 0)
// LDS tiles are stored [row][32 K-halves] so fragment reads are contiguous.
// ---------------------------------------------------------------------------
__device__ __forceinline__ v16h load_frag_a(const _Float16* __restrict__ Asub, int lane) {
  int row = lane & 15;
  int kb  = (lane & 16) ? 8 : 0;
  const _Float16* p = Asub + row * 32;
  v8h lo = *(const v8h*)(p + kb);        // K = kb .. kb+7
  v8h hi = *(const v8h*)(p + kb + 16);   // K = kb+16 .. kb+23
  v16h a;
#pragma unroll
  for (int i = 0; i < 8; ++i) { a[i] = lo[i]; a[i + 8] = hi[i]; }
  return a;
}

__device__ __forceinline__ v16h load_frag_b(const _Float16* __restrict__ Bsub, int lane) {
  int col = lane & 15;
  int kb  = (lane & 16) ? 8 : 0;
  const _Float16* p = Bsub + col * 32 + kb;
  v8h lo = *(const v8h*)(p);
  v8h hi = *(const v8h*)(p + 16);
  v16h b;
#pragma unroll
  for (int i = 0; i < 8; ++i) { b[i] = lo[i]; b[i + 8] = hi[i]; }
  return b;
}

// Low 32 bits of a flat LDS address are the LDS byte offset (ISA 10.2):
// usable directly as the VDST operand of global_load_async_to_lds_*.
__device__ __forceinline__ unsigned lds_off(const void* p) {
  return (unsigned)(uintptr_t)p;
}

// Per-lane async copy of 16 bytes global -> LDS (tracked on ASYNCcnt).
__device__ __forceinline__ void async_ld_b128(unsigned lds, const void* gaddr) {
  asm volatile("global_load_async_to_lds_b128 %0, %1, off"
               :: "v"(lds), "v"(gaddr) : "memory");
}
__device__ __forceinline__ void wait_asynccnt0() {
  asm volatile("s_wait_asynccnt 0x0" ::: "memory");
}

// ---------------------------------------------------------------------------
// Kernel 1: 1x1 conv as GEMM  Y[t][n][o][p] = sum_c W[o][c] * X[n][c][p] + b[o]
// Block = 128 thr (4 waves). Block tile: 64 (o) x 64 (p), K-step 32.
// Register-prefetch pipeline; clamped (non-divergent) row guard + zero-fill.
// grid = (64 p-tiles, 3 m-tiles, 4 = t*2+n)
// ---------------------------------------------------------------------------
__global__ void __launch_bounds__(128)
conv_gemm(const float* __restrict__ fa, const float* __restrict__ fb,
          const float* __restrict__ Wa, const float* __restrict__ ba,
          const float* __restrict__ Wb, const float* __restrict__ bb,
          float* __restrict__ Y)
{
  const int t = blockIdx.z >> 1, n = blockIdx.z & 1;
  const float* __restrict__ X = (t ? fb : fa) + (size_t)n * CIN * HW;
  const float* __restrict__ W = t ? Wb : Wa;
  const float* __restrict__ bias = t ? bb : ba;
  float* __restrict__ Yo = Y + (size_t)(t * 2 + n) * CMID * HW;

  const int m0 = blockIdx.y * 64;
  const int p0 = blockIdx.x * 64;
  const int tid = threadIdx.x, warp = tid >> 5, lane = tid & 31;

  __shared__ __align__(16) _Float16 As[64 * 32];   // [o_local][c_local]
  __shared__ __align__(16) _Float16 Bs[64 * 32];   // [p_local][c_local] (transposed)

  const int p  = p0 + (tid & 63);
  const int cb = (tid >> 6) * 16;

  float4 aPref[4];
  float  bPref[16];

  auto load_stage = [&](int k0) {
#pragma unroll
    for (int it = 0; it < 4; ++it) {
      int id = tid + it * 128;
      int r  = id >> 3, cg = (id & 7) * 4;
      int gr = m0 + r;
      int grc = gr < CMID ? gr : CMID - 1;       // clamped: no exec divergence
      aPref[it] = *(const float4*)(W + (size_t)grc * CIN + k0 + cg);
    }
#pragma unroll
    for (int i = 0; i < 16; ++i)
      bPref[i] = X[(size_t)(k0 + cb + i) * HW + p];
  };

  auto store_stage = [&]() {
#pragma unroll
    for (int it = 0; it < 4; ++it) {
      int id = tid + it * 128;
      int r  = id >> 3, cg = (id & 7) * 4;
      float z = (m0 + r) < CMID ? 1.0f : 0.0f;   // zero pad rows, no branch
      v4h h4;
      h4[0] = (_Float16)(aPref[it].x * z); h4[1] = (_Float16)(aPref[it].y * z);
      h4[2] = (_Float16)(aPref[it].z * z); h4[3] = (_Float16)(aPref[it].w * z);
      *(v4h*)&As[r * 32 + cg] = h4;
    }
#pragma unroll
    for (int h = 0; h < 2; ++h) {
      v8h tmp;
#pragma unroll
      for (int i = 0; i < 8; ++i) tmp[i] = (_Float16)bPref[h * 8 + i];
      *(v8h*)&Bs[(tid & 63) * 32 + cb + h * 8] = tmp;
    }
  };

  v8f acc[4] = {};
  load_stage(0);

  for (int k0 = 0; k0 < CIN; k0 += 32) {
    __syncthreads();
    store_stage();
    __syncthreads();
    if (k0 + 32 < CIN) load_stage(k0 + 32);      // overlaps with WMMA below

    v16h afrag = load_frag_a(&As[warp * 16 * 32], lane);
#pragma unroll
    for (int j = 0; j < 4; ++j) {
      v16h bfrag = load_frag_b(&Bs[j * 16 * 32], lane);
      acc[j] = __builtin_amdgcn_wmma_f32_16x16x32_f16(
          false, afrag, false, bfrag, (short)0, acc[j], false, false);
    }
  }

  const int col = lane & 15, rb = (lane & 16) ? 8 : 0;
#pragma unroll
  for (int j = 0; j < 4; ++j) {
    int gc = p0 + j * 16 + col;
#pragma unroll
    for (int v = 0; v < 8; ++v) {
      int gr = m0 + warp * 16 + rb + v;
      if (gr < CMID) Yo[(size_t)gr * HW + gc] = acc[j][v] + bias[gr];
    }
  }
}

// ---------------------------------------------------------------------------
// Kernel 2: per (t,n,o) mean + rstd over 4096 spatial values (biased var)
// ---------------------------------------------------------------------------
__global__ void __launch_bounds__(256)
stats1(const float* __restrict__ Y, float* __restrict__ mean, float* __restrict__ rstd)
{
  const int id = blockIdx.x;                 // (t*2+n)*136 + o
  const float* __restrict__ y = Y + (size_t)id * HW;
  float s = 0.f, ss = 0.f;
  for (int i = threadIdx.x; i < HW; i += 256) { float v = y[i]; s += v; ss += v * v; }
  __shared__ float rs[256], rq[256];
  rs[threadIdx.x] = s; rq[threadIdx.x] = ss;
  __syncthreads();
  for (int o = 128; o; o >>= 1) {
    if (threadIdx.x < o) { rs[threadIdx.x] += rs[threadIdx.x + o]; rq[threadIdx.x] += rq[threadIdx.x + o]; }
    __syncthreads();
  }
  if (threadIdx.x == 0) {
    float mu  = rs[0] * (1.0f / HW);
    float var = rq[0] * (1.0f / HW) - mu * mu;
    mean[id] = mu;
    rstd[id] = rsqrtf(var + EPS_IN);
  }
}

// ---------------------------------------------------------------------------
// Kernel 3: z = LeakyReLU((y-mu)*rstd) in place; also spatial mean of z
// ---------------------------------------------------------------------------
__global__ void __launch_bounds__(256)
apply_lrelu(float* __restrict__ Y, const float* __restrict__ mean,
            const float* __restrict__ rstd, float* __restrict__ mean2)
{
  const int id = blockIdx.x;
  const float mu = mean[id], rs_ = rstd[id];
  float* __restrict__ y = Y + (size_t)id * HW;
  float s = 0.f;
  for (int i = threadIdx.x; i < HW; i += 256) {
    float v = (y[i] - mu) * rs_;
    v = (v >= 0.f) ? v : SLOPE * v;
    y[i] = v;
    s += v;
  }
  __shared__ float rs[256];
  rs[threadIdx.x] = s;
  __syncthreads();
  for (int o = 128; o; o >>= 1) {
    if (threadIdx.x < o) rs[threadIdx.x] += rs[threadIdx.x + o];
    __syncthreads();
  }
  if (threadIdx.x == 0) mean2[id] = rs[0] * (1.0f / HW);
}

// ---------------------------------------------------------------------------
// Kernel 4: per-pixel channel L2 normalize; emit f16 features as
// flc[t*2+n][pixel][CP] (channel pad 136->160 with zeros)
// ---------------------------------------------------------------------------
__global__ void __launch_bounds__(256)
normalize_pack(const float* __restrict__ Y, const float* __restrict__ mean2,
               _Float16* __restrict__ flc)
{
  const int tn = blockIdx.y;
  const int l  = blockIdx.x * 256 + threadIdx.x;
  const float* __restrict__ z  = Y + (size_t)tn * CMID * HW;
  const float* __restrict__ m2 = mean2 + tn * CMID;
  float s = 0.f;
  for (int c = 0; c < CMID; ++c) {
    float v = z[(size_t)c * HW + l] - m2[c];
    s += v * v;
  }
  float inv = rsqrtf(fmaxf(s, 1e-30f));
  _Float16* __restrict__ dst = flc + ((size_t)tn * HW + l) * CP;
  for (int c = 0; c < CMID; ++c)
    dst[c] = (_Float16)((z[(size_t)c * HW + l] - m2[c]) * inv);
  for (int c = CMID; c < CP; ++c) dst[c] = (_Float16)0.f;
}

// ---------------------------------------------------------------------------
// Kernel 5: energy GEMM. E[k][l] = sum_c A_lc[k][c] * B_lc[l][c], K = CP.
// dir0: A = fb, B = fa (energy_ab_T); dir1: A = fa, B = fb (energy_ba_T).
// Sources are already f16 -> stage with GLOBAL_LOAD_ASYNC_TO_LDS_B128
// (VGPR-free global->LDS, ASYNCcnt-tracked) into double-buffered LDS tiles;
// the async copies of tile c+1 land under the WMMAs of tile c.
// Block = 256 thr (8 waves). Tile 128 (k) x 64 (l). Writes ALPHA*E into d_out.
// grid = (64 l-tiles, 32 k-tiles, 4 = dir*2+n)
// ---------------------------------------------------------------------------
__global__ void __launch_bounds__(256)
energy_gemm(const _Float16* __restrict__ flc, float* __restrict__ corr)
{
  const int dir = blockIdx.z >> 1, n = blockIdx.z & 1;
  const int ta = (dir == 0) ? 1 : 0;       // A-tensor: fb for ab, fa for ba
  const int tb = 1 - ta;
  const _Float16* __restrict__ Alc = flc + (size_t)(ta * 2 + n) * HW * CP;
  const _Float16* __restrict__ Blc = flc + (size_t)(tb * 2 + n) * HW * CP;
  float* __restrict__ C = corr + (size_t)(dir * 2 + n) * HW * HW;

  const int k0r = blockIdx.y * 128;
  const int l0  = blockIdx.x * 64;
  const int tid = threadIdx.x, warp = tid >> 5, lane = tid & 31;

  __shared__ __align__(16) _Float16 As[2][128 * 32];   // [k_local][c_local], x2 buffers
  __shared__ __align__(16) _Float16 Bs[2][64 * 32];    // [l_local][c_local], x2 buffers

  // per-thread 16B chunk assignment (matches fragment layout expectations)
  const int r  = tid >> 2;          // 0..63
  const int off = (tid & 3) * 8;    // 0,8,16,24 halves
  const _Float16* gA0 = Alc + (size_t)(k0r + r)      * CP + off;  // rows 0..63
  const _Float16* gA1 = Alc + (size_t)(k0r + 64 + r) * CP + off;  // rows 64..127
  const _Float16* gB  = Blc + (size_t)(l0 + r)       * CP + off;

  auto issue_async = [&](int c0, int buf) {
    async_ld_b128(lds_off(&As[buf][r * 32 + off]),        gA0 + c0);
    async_ld_b128(lds_off(&As[buf][(64 + r) * 32 + off]), gA1 + c0);
    async_ld_b128(lds_off(&Bs[buf][r * 32 + off]),        gB  + c0);
  };

  v8f acc[4] = {};
  issue_async(0, 0);
  int cur = 0;

  for (int c0 = 0; c0 < CP; c0 += 32) {
    wait_asynccnt0();                 // my chunks for buf[cur] have landed
    __syncthreads();                  // everyone's chunks have landed; prior
                                      // LDS reads of buf[cur^1] are complete
    if (c0 + 32 < CP) issue_async(c0 + 32, cur ^ 1);

    v16h afrag = load_frag_a(&As[cur][warp * 16 * 32], lane);
#pragma unroll
    for (int j = 0; j < 4; ++j) {
      v16h bfrag = load_frag_b(&Bs[cur][j * 16 * 32], lane);
      acc[j] = __builtin_amdgcn_wmma_f32_16x16x32_f16(
          false, afrag, false, bfrag, (short)0, acc[j], false, false);
    }
    cur ^= 1;
  }

  const int col = lane & 15, rb = (lane & 16) ? 8 : 0;
#pragma unroll
  for (int j = 0; j < 4; ++j) {
    int gc = l0 + j * 16 + col;
#pragma unroll
    for (int v = 0; v < 8; ++v) {
      int gr = k0r + warp * 16 + rb + v;
      C[(size_t)gr * HW + gc] = ALPHA * acc[j][v];
    }
  }
}

// ---------------------------------------------------------------------------
// Kernel 6: in-place row softmax; whole 16 KB row cached in LDS so global
// traffic is exactly 1 read + 1 write per element. grid = 16384 rows.
// ---------------------------------------------------------------------------
__global__ void __launch_bounds__(256)
softmax_rows(float* __restrict__ corr)
{
  float* __restrict__ e = corr + (size_t)blockIdx.x * HW;
  const int tid = threadIdx.x;
  __shared__ float rowbuf[HW];          // 16 KB
  __shared__ float red[256];

  float mx = -3.0e38f;
  for (int i = tid; i < HW; i += 256) {
    float v = e[i];
    rowbuf[i] = v;
    mx = fmaxf(mx, v);
  }
  red[tid] = mx;
  __syncthreads();
  for (int o = 128; o; o >>= 1) {
    if (tid < o) red[tid] = fmaxf(red[tid], red[tid + o]);
    __syncthreads();
  }
  mx = red[0];
  __syncthreads();

  float s = 0.f;
  for (int i = tid; i < HW; i += 256) {
    float v = __expf(rowbuf[i] - mx);
    rowbuf[i] = v;
    s += v;
  }
  red[tid] = s;
  __syncthreads();
  for (int o = 128; o; o >>= 1) {
    if (tid < o) red[tid] += red[tid + o];
    __syncthreads();
  }
  float inv = 1.0f / red[0];
  __syncthreads();

  for (int i = tid; i < HW; i += 256) e[i] = rowbuf[i] * inv;
}

// ---------------------------------------------------------------------------
// Kernel 7: warp GEMM. out[c][l] = sum_k raw[c][k] * corr[k][l].  K = 4096.
// Block = 256 thr (8 waves). Tile 128 (c) x 64 (l): corr re-read multiplicity
// 9x (HBM-dominant term). f32 sources need conversion, so staging keeps the
// register-prefetch pipeline. Tail m-tile handled with clamped loads.
// dir0 uses b_raw -> b_warp; dir1 uses a_raw -> a_warp.
// grid = (64 l-tiles, 9 m-tiles, 4 = dir*2+n)
// ---------------------------------------------------------------------------
__global__ void __launch_bounds__(256)
warp_gemm(const float* __restrict__ a_raw, const float* __restrict__ b_raw,
          const float* __restrict__ corr, float* __restrict__ warp_out)
{
  const int dir = blockIdx.z >> 1, n = blockIdx.z & 1;
  const float* __restrict__ R  = ((dir == 0) ? b_raw : a_raw) + (size_t)n * CIN * HW;
  const float* __restrict__ Cr = corr + (size_t)(dir * 2 + n) * HW * HW;
  float* __restrict__ O = warp_out
      + ((dir == 0) ? (size_t)2 * CIN * HW : (size_t)0)   // b_warp after a_warp
      + (size_t)n * CIN * HW;

  const int m0 = blockIdx.y * 128;
  const int l0 = blockIdx.x * 64;
  const int tid = threadIdx.x, warp = tid >> 5, lane = tid & 31;

  __shared__ __align__(16) _Float16 As[128 * 32];   // [c_local][k_local]
  __shared__ __align__(16) _Float16 Bs[64 * 32];    // [l_local][k_local] (transposed)

  const int ar = tid >> 1;               // 0..127  (A row)
  const int ako = (tid & 1) * 16;        // 0 / 16
  const int gr_a = m0 + ar;
  const int gra_c = gr_a < CIN ? gr_a : CIN - 1;    // clamped, non-divergent
  const float za = gr_a < CIN ? 1.0f : 0.0f;

  const int bkl = tid >> 3;              // 0..31   (B k row)
  const int blg = (tid & 7) * 8;         // 0,8,..,56

  float4 aPref[4];
  float4 bPref[2];

  auto load_stage = [&](int k0) {
    const float* sa = R + (size_t)gra_c * HW + k0 + ako;
#pragma unroll
    for (int q = 0; q < 4; ++q) aPref[q] = *(const float4*)(sa + q * 4);
    const float* sb = Cr + (size_t)(k0 + bkl) * HW + l0 + blg;
    bPref[0] = *(const float4*)(sb);
    bPref[1] = *(const float4*)(sb + 4);
  };
  auto store_stage = [&]() {
#pragma unroll
    for (int h = 0; h < 2; ++h) {
      v8h tmp;
      tmp[0] = (_Float16)(aPref[h*2].x * za);   tmp[1] = (_Float16)(aPref[h*2].y * za);
      tmp[2] = (_Float16)(aPref[h*2].z * za);   tmp[3] = (_Float16)(aPref[h*2].w * za);
      tmp[4] = (_Float16)(aPref[h*2+1].x * za); tmp[5] = (_Float16)(aPref[h*2+1].y * za);
      tmp[6] = (_Float16)(aPref[h*2+1].z * za); tmp[7] = (_Float16)(aPref[h*2+1].w * za);
      *(v8h*)&As[ar * 32 + ako + h * 8] = tmp;
    }
    const float* bp = (const float*)&bPref[0];
#pragma unroll
    for (int i = 0; i < 8; ++i)
      Bs[(blg + i) * 32 + bkl] = (_Float16)bp[i];
  };

  v8f acc[4] = {};
  load_stage(0);

  for (int k0 = 0; k0 < HW; k0 += 32) {
    __syncthreads();
    store_stage();
    __syncthreads();
    if (k0 + 32 < HW) load_stage(k0 + 32);      // loads hide under WMMAs

    v16h afrag = load_frag_a(&As[warp * 16 * 32], lane);
#pragma unroll
    for (int j = 0; j < 4; ++j) {
      v16h bfrag = load_frag_b(&Bs[j * 16 * 32], lane);
      acc[j] = __builtin_amdgcn_wmma_f32_16x16x32_f16(
          false, afrag, false, bfrag, (short)0, acc[j], false, false);
    }
  }

  const int col = lane & 15, rb = (lane & 16) ? 8 : 0;
#pragma unroll
  for (int j = 0; j < 4; ++j) {
    int gc = l0 + j * 16 + col;
#pragma unroll
    for (int v = 0; v < 8; ++v) {
      int gr = m0 + warp * 16 + rb + v;
      if (gr < CIN) O[(size_t)gr * HW + gc] = acc[j][v];
    }
  }
}

// ---------------------------------------------------------------------------
// Host launcher
// ---------------------------------------------------------------------------
extern "C" void kernel_launch(void* const* d_in, const int* in_sizes, int n_in,
                              void* d_out, int out_size, void* d_ws, size_t ws_size,
                              hipStream_t stream) {
  (void)in_sizes; (void)n_in; (void)out_size; (void)ws_size;
  const float* fa    = (const float*)d_in[0];
  const float* fb    = (const float*)d_in[1];
  const float* a_raw = (const float*)d_in[2];
  const float* b_raw = (const float*)d_in[3];
  const float* Wa    = (const float*)d_in[4];
  const float* ba    = (const float*)d_in[5];
  const float* Wb    = (const float*)d_in[6];
  const float* bb    = (const float*)d_in[7];

  float* out  = (float*)d_out;
  float* corr = out;                                // corr_ab_T then corr_ba_T
  float* warp = out + (size_t)4 * HW * HW;          // a_warp then b_warp

  // workspace layout (~14 MB)
  float* Y      = (float*)d_ws;                     // [2][2][136][4096] f32
  float* mean_  = Y + (size_t)4 * CMID * HW;
  float* rstd_  = mean_ + 4 * CMID;
  float* mean2_ = rstd_ + 4 * CMID;
  _Float16* flc = (_Float16*)(mean2_ + 4 * CMID);   // [2][2][4096][160] f16

  conv_gemm     <<<dim3(64, 3, 4),  128, 0, stream>>>(fa, fb, Wa, ba, Wb, bb, Y);
  stats1        <<<dim3(4 * CMID),  256, 0, stream>>>(Y, mean_, rstd_);
  apply_lrelu   <<<dim3(4 * CMID),  256, 0, stream>>>(Y, mean_, rstd_, mean2_);
  normalize_pack<<<dim3(16, 4),     256, 0, stream>>>(Y, mean2_, flc);
  energy_gemm   <<<dim3(64, 32, 4), 256, 0, stream>>>(flc, corr);
  softmax_rows  <<<dim3(4 * HW),    256, 0, stream>>>(corr);
  warp_gemm     <<<dim3(64, 9, 4),  256, 0, stream>>>(a_raw, b_raw, corr, warp);
}
